// AnnotatedGraphTripleConv_60232621359657
// MI455X (gfx1250) — matrix operand order repaired and checked by reading, not compiled, not run on hardware
//
#include <hip/hip_runtime.h>

typedef __bf16 bf16;
typedef __attribute__((ext_vector_type(16))) __bf16 v16bf;
typedef __attribute__((ext_vector_type(8)))  __bf16 v8bf;
typedef __attribute__((ext_vector_type(8)))  float  v8f;

static __device__ __forceinline__ unsigned pk2(float a, float b) {
    unsigned short ua = __builtin_bit_cast(unsigned short, (__bf16)a);
    unsigned short ub = __builtin_bit_cast(unsigned short, (__bf16)b);
    return (unsigned)ua | ((unsigned)ub << 16);
}

// Load a 16x32 bf16 A-fragment from an LDS tile (row-major, `stride` bf16 elems/row).
// Wave32 A layout: lane L holds row M=L&15; element e holds K = kstep*32 + (e<8?e:e+8) + (L>>4)*8.
static __device__ __forceinline__ v16bf load_af(const bf16* buf, int row, int stride, int k, int lh) {
    const bf16* p = buf + row * stride + k * 32 + lh * 8;
    v8bf lo = *(const v8bf*)(p);
    v8bf hi = *(const v8bf*)(p + 16);
    return __builtin_shufflevector(lo, hi, 0,1,2,3,4,5,6,7,8,9,10,11,12,13,14,15);
}

// ---------------------------------------------------------------------------
// Repack fp32 weight matrix [K][N] (row-major) into bf16 B-fragment layout:
// flat = ((ntile*(K/32)+kstep)*32 + lane)*16 + e
// value = W[kstep*32 + (lane>>4)*8 + (e&7) + (e>>3)*16][ntile*16 + (lane&15)]
// ---------------------------------------------------------------------------
__global__ void pack_w(const float* __restrict__ src, bf16* __restrict__ dst, int K, int N) {
    int i = blockIdx.x * blockDim.x + threadIdx.x;
    if (i >= K * N) return;
    int e    = i & 15;
    int lane = (i >> 4) & 31;
    int ks   = (i >> 9) % (K / 32);
    int nt   = i / ((K / 32) * 512);
    int row  = ks * 32 + (lane >> 4) * 8 + (e & 7) + (e >> 3) * 16;
    int col  = nt * 16 + (lane & 15);
    dst[i] = (bf16)src[row * N + col];
}

// ---------------------------------------------------------------------------
// Fused triple MLP: t=concat(obj[s],pred,obj[o]); h=relu(t@W1a+b1a);
// new_t=h@W1b+b1b; scatter new_s/new_o into pooled (atomics), store new_p.
// One wave handles a 16-triple tile.
// ---------------------------------------------------------------------------
#define NW1 4
__global__ void __launch_bounds__(128) k_triples(
    const float* __restrict__ obj, const float* __restrict__ pred,
    const int* __restrict__ edges,
    const bf16* __restrict__ w1ap, const float* __restrict__ b1a,
    const bf16* __restrict__ w1bp, const float* __restrict__ b1b,
    float* __restrict__ pooled, float* __restrict__ counts,
    float* __restrict__ outP, int T)
{
    __shared__ bf16 tb[NW1][16 * 392];      // t staging (padded stride), reused for h (stride 136)
    __shared__ int  sidx[NW1][16], oidx[NW1][16];

    const int w    = threadIdx.x >> 5;
    const int lane = threadIdx.x & 31;
    const int lh   = lane >> 4;
    const int l16  = lane & 15;
    const int tile = blockIdx.x * NW1 + w;
    if (tile * 16 >= T) return;
    const int base = tile * 16;

    // --- indices + count atomics -------------------------------------------
    if (lane < 16) {
        int tr = base + lane; if (tr > T - 1) tr = T - 1;
        int s = edges[2 * tr], o = edges[2 * tr + 1];
        sidx[w][lane] = s; oidx[w][lane] = o;
        if (base + lane < T) atomicAdd(&counts[s], 1.0f);
    } else {
        int tr = base + lane - 16; if (tr > T - 1) tr = T - 1;
        int o = edges[2 * tr + 1];
        if (base + lane - 16 < T) atomicAdd(&counts[o], 1.0f);
    }

    // --- gather + fp32->bf16 stage of t (16 x 384, stride 392) -------------
    for (int m = 0; m < 16; ++m) {
        int tr = base + m; if (tr > T - 1) tr = T - 1;
        int s = sidx[w][m], o = oidx[w][m];
        float4 vs = *(const float4*)(obj  + (size_t)s  * 128 + lane * 4);
        float4 vp = *(const float4*)(pred + (size_t)tr * 128 + lane * 4);
        float4 vo = *(const float4*)(obj  + (size_t)o  * 128 + lane * 4);
        bf16* row = &tb[w][m * 392];
        *(uint2*)(row +       lane * 4) = make_uint2(pk2(vs.x, vs.y), pk2(vs.z, vs.w));
        *(uint2*)(row + 128 + lane * 4) = make_uint2(pk2(vp.x, vp.y), pk2(vp.z, vp.w));
        *(uint2*)(row + 256 + lane * 4) = make_uint2(pk2(vo.x, vo.y), pk2(vo.z, vo.w));
    }

    // --- A fragments for GEMM1 (K=384 -> 12 ksteps) ------------------------
    v16bf a1[12];
#pragma unroll
    for (int k = 0; k < 12; ++k) a1[k] = load_af(&tb[w][0], l16, 392, k, lh);

    // --- GEMM1 + relu -> h (bf16) in LDS (stride 136) ----------------------
    bf16* hb = &tb[w][0];
#pragma unroll
    for (int n = 0; n < 8; ++n) {
        float bias = b1a[n * 16 + l16];
        v8f c;
#pragma unroll
        for (int r = 0; r < 8; ++r) c[r] = bias;
#pragma unroll
        for (int k = 0; k < 12; ++k) {
            v16bf b = *(const v16bf*)(w1ap + ((size_t)(n * 12 + k) * 32 + lane) * 16);
            c = __builtin_amdgcn_wmma_f32_16x16x32_bf16(false, a1[k], false, b,
                                                        (short)0, c, false, false);
        }
#pragma unroll
        for (int r = 0; r < 8; ++r) {
            float v = fmaxf(c[r], 0.0f);
            hb[(r + lh * 8) * 136 + n * 16 + l16] = (bf16)v;
        }
    }

    // --- A fragments for GEMM2 (K=128 -> 4 ksteps) -------------------------
    v16bf a2[4];
#pragma unroll
    for (int k = 0; k < 4; ++k) a2[k] = load_af(hb, l16, 136, k, lh);

    // --- GEMM2: N = 384 (24 ntiles); route outputs -------------------------
#pragma unroll
    for (int n = 0; n < 24; ++n) {
        float bias = b1b[n * 16 + l16];
        v8f c;
#pragma unroll
        for (int r = 0; r < 8; ++r) c[r] = bias;
#pragma unroll
        for (int k = 0; k < 4; ++k) {
            v16bf b = *(const v16bf*)(w1bp + ((size_t)(n * 4 + k) * 32 + lane) * 16);
            c = __builtin_amdgcn_wmma_f32_16x16x32_bf16(false, a2[k], false, b,
                                                        (short)0, c, false, false);
        }
        int col = n * 16 + l16;
#pragma unroll
        for (int r = 0; r < 8; ++r) {
            int m = r + lh * 8;
            if (base + m >= T) continue;
            float v = c[r];
            if (n < 8) {
                atomicAdd(&pooled[(size_t)sidx[w][m] * 128 + col], v);
            } else if (n < 16) {
                outP[(size_t)(base + m) * 128 + (col - 128)] = v;
            } else {
                atomicAdd(&pooled[(size_t)oidx[w][m] * 128 + (col - 256)], v);
            }
        }
    }
}

// ---------------------------------------------------------------------------
// Object MLP: new_obj = relu((pooled/cnt)@W2a+b2a)@W2b + b2b
// One wave handles a 16-object tile.
// ---------------------------------------------------------------------------
__global__ void __launch_bounds__(128) k_objects(
    const float* __restrict__ pooled, const float* __restrict__ counts,
    const bf16* __restrict__ w2ap, const float* __restrict__ b2a,
    const bf16* __restrict__ w2bp, const float* __restrict__ b2b,
    float* __restrict__ outObj, int O)
{
    __shared__ bf16 ab[NW1][16 * 136];
    const int w    = threadIdx.x >> 5;
    const int lane = threadIdx.x & 31;
    const int lh   = lane >> 4;
    const int l16  = lane & 15;
    const int tile = blockIdx.x * NW1 + w;
    if (tile * 16 >= O) return;
    const int base = tile * 16;

    for (int m = 0; m < 16; ++m) {
        int ob = base + m; if (ob > O - 1) ob = O - 1;
        float inv = 1.0f / fmaxf(counts[ob], 1.0f);
        float4 v = *(const float4*)(pooled + (size_t)ob * 128 + lane * 4);
        *(uint2*)(&ab[w][m * 136] + lane * 4) =
            make_uint2(pk2(v.x * inv, v.y * inv), pk2(v.z * inv, v.w * inv));
    }

    v16bf a[4];
#pragma unroll
    for (int k = 0; k < 4; ++k) a[k] = load_af(&ab[w][0], l16, 136, k, lh);

    // GEMM3 + relu -> back into LDS (A fragments already in registers)
#pragma unroll
    for (int n = 0; n < 8; ++n) {
        float bias = b2a[n * 16 + l16];
        v8f c;
#pragma unroll
        for (int r = 0; r < 8; ++r) c[r] = bias;
#pragma unroll
        for (int k = 0; k < 4; ++k) {
            v16bf b = *(const v16bf*)(w2ap + ((size_t)(n * 4 + k) * 32 + lane) * 16);
            c = __builtin_amdgcn_wmma_f32_16x16x32_bf16(false, a[k], false, b,
                                                        (short)0, c, false, false);
        }
#pragma unroll
        for (int r = 0; r < 8; ++r) {
            float v = fmaxf(c[r], 0.0f);
            ab[w][(r + lh * 8) * 136 + n * 16 + l16] = (bf16)v;
        }
    }

    v16bf a2[4];
#pragma unroll
    for (int k = 0; k < 4; ++k) a2[k] = load_af(&ab[w][0], l16, 136, k, lh);

    // GEMM4 -> new_obj (fp32)
#pragma unroll
    for (int n = 0; n < 8; ++n) {
        float bias = b2b[n * 16 + l16];
        v8f c;
#pragma unroll
        for (int r = 0; r < 8; ++r) c[r] = bias;
#pragma unroll
        for (int k = 0; k < 4; ++k) {
            v16bf b = *(const v16bf*)(w2bp + ((size_t)(n * 4 + k) * 32 + lane) * 16);
            c = __builtin_amdgcn_wmma_f32_16x16x32_bf16(false, a2[k], false, b,
                                                        (short)0, c, false, false);
        }
        int col = n * 16 + l16;
#pragma unroll
        for (int r = 0; r < 8; ++r) {
            int m = r + lh * 8;
            if (base + m < O) outObj[(size_t)(base + m) * 128 + col] = c[r];
        }
    }
}

static inline size_t align256(size_t x) { return (x + 255) & ~(size_t)255; }

extern "C" void kernel_launch(void* const* d_in, const int* in_sizes, int n_in,
                              void* d_out, int out_size, void* d_ws, size_t ws_size,
                              hipStream_t stream) {
    (void)n_in; (void)out_size; (void)ws_size;
    const float* obj  = (const float*)d_in[0];
    const float* pred = (const float*)d_in[1];
    const int*   edges= (const int*)  d_in[2];
    const float* W1a  = (const float*)d_in[3];
    const float* b1a  = (const float*)d_in[4];
    const float* W1b  = (const float*)d_in[5];
    const float* b1b  = (const float*)d_in[6];
    const float* W2a  = (const float*)d_in[7];
    const float* b2a  = (const float*)d_in[8];
    const float* W2b  = (const float*)d_in[9];
    const float* b2b  = (const float*)d_in[10];

    const int O = in_sizes[0] / 128;
    const int T = in_sizes[1] / 128;

    float* out    = (float*)d_out;
    float* outObj = out;                       // (O,128)
    float* outP   = out + (size_t)O * 128;     // (T,128)

    char* ws = (char*)d_ws;
    size_t off = 0;
    float* pooled = (float*)(ws + off); off += align256((size_t)O * 128 * sizeof(float));
    float* counts = (float*)(ws + off); off += align256((size_t)O * sizeof(float));
    const size_t zeroBytes = off;
    bf16* w1ap = (bf16*)(ws + off); off += align256((size_t)384 * 128 * sizeof(bf16));
    bf16* w1bp = (bf16*)(ws + off); off += align256((size_t)128 * 384 * sizeof(bf16));
    bf16* w2ap = (bf16*)(ws + off); off += align256((size_t)128 * 128 * sizeof(bf16));
    bf16* w2bp = (bf16*)(ws + off); off += align256((size_t)128 * 128 * sizeof(bf16));

    hipMemsetAsync(ws, 0, zeroBytes, stream);

    pack_w<<<(384 * 128 + 255) / 256, 256, 0, stream>>>(W1a, w1ap, 384, 128);
    pack_w<<<(128 * 384 + 255) / 256, 256, 0, stream>>>(W1b, w1bp, 128, 384);
    pack_w<<<(128 * 128 + 255) / 256, 256, 0, stream>>>(W2a, w2ap, 128, 128);
    pack_w<<<(128 * 128 + 255) / 256, 256, 0, stream>>>(W2b, w2bp, 128, 128);

    const int tiles1 = (T + 15) / 16;
    const int tiles3 = (O + 15) / 16;
    k_triples<<<(tiles1 + NW1 - 1) / NW1, 128, 0, stream>>>(
        obj, pred, edges, w1ap, b1a, w1bp, b1b, pooled, counts, outP, T);
    k_objects<<<(tiles3 + NW1 - 1) / NW1, 128, 0, stream>>>(
        pooled, counts, w2ap, b2a, w2bp, b2b, outObj, O);
}
